// Model_39676907882035
// MI455X (gfx1250) — matrix operand order repaired and checked by reading, not compiled, not run on hardware
//
#include <hip/hip_runtime.h>

typedef __attribute__((ext_vector_type(16))) _Float16 v16h;
typedef __attribute__((ext_vector_type(8)))  float    v8f;

namespace {
constexpr int      kS       = 1415;
constexpr int      kD       = 120;
constexpr float    kKeep    = 0.2539835287718966f;   // 1 - P_DROP
constexpr unsigned kNHalf   = 8008900u;              // (8*S*S)/2 == 4*S*S
constexpr float    kScale   = 0.09128709291752769f;  // 1/sqrt(120)
constexpr float    kNegBig  = -1.0e30f;
constexpr int      kWaves   = 4;                     // key-split waves per workgroup
constexpr unsigned kMaskWords  = (kNHalf + 31u) / 32u;  // 250279
constexpr unsigned kMaskStride = kMaskWords + 1u;       // +1 guard word for w+1 reads
}

__device__ __forceinline__ unsigned rotl32(unsigned x, unsigned n) {
  return (x << n) | (x >> (32u - n));
}

// JAX threefry2x32 with key = (0, 42)   [jax.random.key(42)]
__device__ __forceinline__ void threefry2x32_42(unsigned& x0, unsigned& x1) {
  const unsigned ks0 = 0u, ks1 = 42u;
  const unsigned ks2 = ks0 ^ ks1 ^ 0x1BD11BDAu;
  x0 += ks0; x1 += ks1;
#define TF_R4(a,b,c,d)                         \
  { x0 += x1; x1 = rotl32(x1,(a)); x1 ^= x0;   \
    x0 += x1; x1 = rotl32(x1,(b)); x1 ^= x0;   \
    x0 += x1; x1 = rotl32(x1,(c)); x1 ^= x0;   \
    x0 += x1; x1 = rotl32(x1,(d)); x1 ^= x0; }
  TF_R4(13,15,26,6)  x0 += ks1; x1 += ks2 + 1u;
  TF_R4(17,29,16,24) x0 += ks2; x1 += ks0 + 2u;
  TF_R4(13,15,26,6)  x0 += ks0; x1 += ks1 + 3u;
  TF_R4(17,29,16,24) x0 += ks1; x1 += ks2 + 4u;
  TF_R4(13,15,26,6)  x0 += ks2; x1 += ks0 + 5u;
#undef TF_R4
}

// ---- Pass 1: precompute dropout keep-bit mask (1 bit / element, lo & hi halves) ----
__global__ __launch_bounds__(256)
void dropmask_kernel(unsigned* __restrict__ lo_words, unsigned* __restrict__ hi_words) {
  const unsigned t = blockIdx.x * 256u + threadIdx.x;
  if (t >= kMaskStride) return;
  unsigned lo = 0u, hi = 0u;
#pragma unroll 4
  for (unsigned b = 0u; b < 32u; ++b) {
    unsigned x0 = t * 32u + b;
    unsigned x1 = x0 + kNHalf;
    threefry2x32_42(x0, x1);
    const unsigned klo = ((__uint_as_float((x0 >> 9) | 0x3f800000u) - 1.0f) < kKeep) ? 1u : 0u;
    const unsigned khi = ((__uint_as_float((x1 >> 9) | 0x3f800000u) - 1.0f) < kKeep) ? 1u : 0u;
    lo |= klo << b;
    hi |= khi << b;
  }
  lo_words[t] = lo;
  hi_words[t] = hi;
}

// ---- Pass 2: flash attention, one head per wave, 4-way key-split per workgroup ----
__global__ __launch_bounds__(32 * kWaves)
__attribute__((amdgpu_waves_per_eu(4)))     // <=256 VGPRs: no s_set_vgpr_msb tax
void fattn_wmma_kernel(const float* __restrict__ Qp, const float* __restrict__ Kp,
                       const float* __restrict__ Vp, const float* __restrict__ Mp,
                       float* __restrict__ Op,
                       const unsigned* __restrict__ lo_words,
                       const unsigned* __restrict__ hi_words) {
  __shared__ _Float16 pbuf[kWaves][16 * 32];   // 4 KB: per-wave P staging
  __shared__ float    obuf[16][128];           // 8 KB: cross-wave combine
  __shared__ float    mbuf[kWaves][16];
  __shared__ float    lbuf[kWaves][16];

  const int tid  = threadIdx.x;
  const int wave = tid >> 5;
  const int lane = tid & 31;
  const int r    = lane & 15;
  const int hh   = lane >> 4;          // lane half
  const int q0   = blockIdx.x * 16;    // query tile
  const int head = blockIdx.y;         // 0..7
  const int headm = head & 3;
  const unsigned* __restrict__ mk = (head < 4) ? lo_words : hi_words;

  // bit index of (headm, clamped tile top, col 0); rows step by S bits
  const int qtop = (q0 < kS - 15) ? q0 : (kS - 16);
  const unsigned erow0 = ((unsigned)headm * (unsigned)kS + (unsigned)qtop) * (unsigned)kS;

  // zero the combine accumulator (ordering covered by post-loop barrier)
  for (int i = tid; i < 16 * 128; i += 32 * kWaves)
    (&obuf[0][0])[i] = 0.f;

  // ---- Load Q tile into A-operands (branchless, clamped), pre-scaled by 1/sqrt(d) ----
  const int  qrow  = q0 + r;
  const bool qok   = qrow < kS;
  const int  qrowc = qok ? qrow : (kS - 1);
  const float* qbase = Qp + (size_t)(head * kS + qrowc) * kD;
  v16h qa[4];
#pragma unroll
  for (int c = 0; c < 4; ++c) {
#pragma unroll
    for (int g = 0; g < 4; ++g) {                 // 4 contiguous float4 runs per chunk
      const int eb  = 4 * g;
      const int d0  = 32 * c + ((eb >> 3) << 4) + (hh << 3) + (eb & 7);  // A layout
      const int d0c = (d0 < kD) ? d0 : (kD - 4);  // clamp: in-row, 16B aligned
      const float4 p4 = *reinterpret_cast<const float4*>(qbase + d0c);
      const bool v = qok && (d0 < kD);
      qa[c][eb + 0] = (_Float16)(v ? p4.x * kScale : 0.f);
      qa[c][eb + 1] = (_Float16)(v ? p4.y * kScale : 0.f);
      qa[c][eb + 2] = (_Float16)(v ? p4.z * kScale : 0.f);
      qa[c][eb + 3] = (_Float16)(v ? p4.w * kScale : 0.f);
    }
  }

  // ---- Online-softmax state ----
  float m_run[8], l_run[8];
  v8f   acc[8];
#pragma unroll
  for (int j = 0; j < 8; ++j) { m_run[j] = -3.0e38f; l_run[j] = 0.f; }
#pragma unroll
  for (int t = 0; t < 8; ++t) acc[t] = v8f{};

  // ---- Key loop: this wave owns key blocks k0 = 32*(wave + 4*i) ----
  for (int k0 = 32 * wave; k0 < kS; k0 += 32 * kWaves) {
    if (k0 + 32 * kWaves < kS)
      __builtin_prefetch(Kp + (size_t)(head * kS + k0 + 32 * kWaves + r) * kD, 0, 0);

    // -- scores: two 16x16 f32 tiles over the 32-key block --
    v8f sc[2];
#pragma unroll
    for (int t = 0; t < 2; ++t) {
      const int  krow  = k0 + 16 * t + r;
      const bool kok   = krow < kS;
      const int  krowc = kok ? krow : (kS - 1);
      const float* kbase = Kp + (size_t)(head * kS + krowc) * kD;
      v16h kb[4];
#pragma unroll
      for (int c = 0; c < 4; ++c) {
        const int dbase = 32 * c + (hh << 3);     // B layout: table order (8*hh + 16*(e>>3) + (e&7))
#pragma unroll
        for (int qd = 0; qd < 4; ++qd) {
          const int d0  = dbase + ((qd >> 1) << 4) + 4 * (qd & 1);
          const int d0c = (d0 < kD) ? d0 : (kD - 4);
          const float4 p4 = *reinterpret_cast<const float4*>(kbase + d0c);
          const bool v = kok && (d0 < kD);
          kb[c][4 * qd + 0] = (_Float16)(v ? p4.x : 0.f);
          kb[c][4 * qd + 1] = (_Float16)(v ? p4.y : 0.f);
          kb[c][4 * qd + 2] = (_Float16)(v ? p4.z : 0.f);
          kb[c][4 * qd + 3] = (_Float16)(v ? p4.w : 0.f);
        }
      }
      v8f cacc = v8f{};
#pragma unroll
      for (int c = 0; c < 4; ++c)
        cacc = __builtin_amdgcn_wmma_f32_16x16x32_f16(false, qa[c], false, kb[c],
                                                      (short)0, cacc, false, false);
      sc[t] = cacc;
    }

    // -- add mask (branchless clamped load + select), bounds -> -inf --
#pragma unroll
    for (int t = 0; t < 2; ++t) {
      const int kcol  = k0 + 16 * t + r;          // C layout: col = lane&15
      const int kcolc = (kcol < kS) ? kcol : (kS - 1);
#pragma unroll
      for (int j = 0; j < 8; ++j) {
        const int  qr  = q0 + j + (hh << 3);      // C layout: row = vgpr + 8*half
        const int  qrc = (qr < kS) ? qr : (kS - 1);
        const bool ok  = (kcol < kS) && (qr < kS);
        const float mv = Mp[(size_t)qrc * kS + kcolc];
        sc[t][j] = ok ? (sc[t][j] + mv) : kNegBig;
      }
    }

    // -- online softmax + bitmask dropout; stage P (f16) to LDS --
    float alpha[8];
#pragma unroll
    for (int j = 0; j < 8; ++j) {
      // issue dropout-mask window loads first so they overlap the reductions
      const unsigned ebase = erow0 + (unsigned)(j + (hh << 3)) * (unsigned)kS + (unsigned)k0;
      const unsigned w     = ebase >> 5;
      const unsigned off   = ebase & 31u;
      const unsigned wlo   = mk[w];
      const unsigned whi   = mk[w + 1];

      float bm = fmaxf(sc[0][j], sc[1][j]);
#pragma unroll
      for (int off2 = 8; off2 >= 1; off2 >>= 1)
        bm = fmaxf(bm, __shfl_xor(bm, off2, 32));   // reduce within 16-lane half
      const float mn = fmaxf(m_run[j], bm);
      const float al = __expf(m_run[j] - mn);
      float p0 = __expf(sc[0][j] - mn);
      float p1 = __expf(sc[1][j] - mn);
      float rs = p0 + p1;
#pragma unroll
      for (int off2 = 8; off2 >= 1; off2 >>= 1)
        rs += __shfl_xor(rs, off2, 32);
      l_run[j] = l_run[j] * al + rs;   // denominator: pre-dropout weights
      m_run[j] = mn;
      alpha[j] = al;

      const unsigned long long win =
          ((unsigned long long)whi << 32) | (unsigned long long)wlo;
      if (((win >> (off + (unsigned)r)) & 1ull) == 0ull)       p0 = 0.f;
      if (((win >> (off + 16u + (unsigned)r)) & 1ull) == 0ull) p1 = 0.f;

      pbuf[wave][(j + (hh << 3)) * 32 + r]      = (_Float16)p0;
      pbuf[wave][(j + (hh << 3)) * 32 + 16 + r] = (_Float16)p1;
    }

    // -- reload P in A-operand layout (in-wave ds ordering is safe) --
    v16h pa;
#pragma unroll
    for (int g = 0; g < 8; ++g) {
      const int e  = 2 * g;
      const int kk = ((e >> 3) << 4) + (hh << 3) + (e & 7);
      pa[e]     = pbuf[wave][r * 32 + kk];
      pa[e + 1] = pbuf[wave][r * 32 + kk + 1];
    }

    // -- O += P @ V over eight 16-wide d tiles (branchless clamped gather) --
    const float* vbase = Vp + (size_t)head * kS * kD;
#pragma unroll
    for (int dt = 0; dt < 8; ++dt) {
      const int  dcol  = dt * 16 + r;             // B layout: col = lane&15
      const bool dok   = (dt < 7) || (dcol < kD);
      const int  dcolc = dok ? dcol : (kD - 1);
      v16h vb;
#pragma unroll
      for (int e = 0; e < 16; ++e) {
        const int krow  = k0 + ((e >> 3) << 4) + (hh << 3) + (e & 7);     // B layout: table order
        const int krowc = (krow < kS) ? krow : (kS - 1);
        const float x = vbase[(size_t)krowc * kD + dcolc];
        vb[e] = (_Float16)((dok && krow < kS) ? x : 0.f);
      }
      v8f a = acc[dt];
#pragma unroll
      for (int j = 0; j < 8; ++j) a[j] *= alpha[j];  // same lane pattern as C
      acc[dt] = __builtin_amdgcn_wmma_f32_16x16x32_f16(false, pa, false, vb,
                                                       (short)0, a, false, false);
    }
  }

  // ---- Cross-wave combine of partial flash states ----
  if (r == 0) {
#pragma unroll
    for (int j = 0; j < 8; ++j) {
      mbuf[wave][j + (hh << 3)] = m_run[j];
      lbuf[wave][j + (hh << 3)] = l_run[j];
    }
  }
  __syncthreads();

#pragma unroll
  for (int j = 0; j < 8; ++j) {
    const int row = j + (hh << 3);
    float mt = mbuf[0][row];
#pragma unroll
    for (int w = 1; w < kWaves; ++w) mt = fmaxf(mt, mbuf[w][row]);
    const float as = __expf(m_run[j] - mt);
#pragma unroll
    for (int dt = 0; dt < 8; ++dt)
      atomicAdd(&obuf[row][dt * 16 + r], acc[dt][j] * as);
  }
  __syncthreads();

  // ---- Finalize: O = sum / (l_tot * keep) ----
  for (int i = tid; i < 16 * kD; i += 32 * kWaves) {
    const int row = i / kD;
    const int col = i - row * kD;
    const int qr  = q0 + row;
    if (qr >= kS) continue;
    float mt = mbuf[0][row];
#pragma unroll
    for (int w = 1; w < kWaves; ++w) mt = fmaxf(mt, mbuf[w][row]);
    float lt = 0.f;
#pragma unroll
    for (int w = 0; w < kWaves; ++w)
      lt += __expf(mbuf[w][row] - mt) * lbuf[w][row];
    Op[(size_t)(head * kS + qr) * kD + col] = obuf[row][col] / (lt * kKeep);
  }
}

extern "C" void kernel_launch(void* const* d_in, const int* in_sizes, int n_in,
                              void* d_out, int out_size, void* d_ws, size_t ws_size,
                              hipStream_t stream) {
  (void)in_sizes; (void)n_in; (void)out_size; (void)ws_size;
  const float* Q = (const float*)d_in[0];
  const float* K = (const float*)d_in[1];
  const float* V = (const float*)d_in[2];
  const float* M = (const float*)d_in[3];
  float* O = (float*)d_out;

  unsigned* lo = (unsigned*)d_ws;             // ~1 MB low-half keep bits
  unsigned* hi = lo + kMaskStride;            // ~1 MB high-half keep bits

  dropmask_kernel<<<(kMaskStride + 255u) / 256u, 256, 0, stream>>>(lo, hi);

  dim3 grid((kS + 15) / 16, 8);               // 89 query tiles x 8 heads
  fattn_wmma_kernel<<<grid, 32 * kWaves, 0, stream>>>(Q, K, V, M, O, lo, hi);
}
